// DispatchEinsumCombine_62878321214333
// MI455X (gfx1250) — matrix operand order repaired and checked by reading, not compiled, not run on hardware
//
#include <hip/hip_runtime.h>
#include <hip/hip_bf16.h>

#define S_TOK 2048
#define HDIM 1024
#define NEXP 8
#define INTER 2048
#define TWOI 4096
#define ALPHA 1.702f
#define LIMIT 7.0f

// expert segments padded to 64 rows: 4096 assignments + 8*63 pad <= 4608
#define SLOT_CAP 4608
#define TILES_M (SLOT_CAP / 64)   // 72

typedef __attribute__((ext_vector_type(16))) __bf16 v16bf;
typedef __attribute__((ext_vector_type(8))) float v8f;
typedef __attribute__((ext_vector_type(8))) unsigned int v8u;

// ---- workspace layout (bytes) ----
#define OFF_TOPK_IDX 0              // int[S*2]
#define OFF_TOPK_W   16384          // float[S*2]
#define OFF_SLOTOF   32768          // int[S*2]
#define OFF_COUNTS   49152          // int[8]
#define OFF_COUNTS2  49184          // int[8]
#define OFF_OFFS     49216          // int[9] (padded)
#define OFF_ETOK     49280          // int[SLOT_CAP]
#define OFF_XB       67712          // bf16[S*H]                  (4 MB)
#define OFF_ACT      4262016        // bf16[SLOT_CAP*INTER]       (18.9 MB)
#define OFF_DOUT     23136384       // f32 [SLOT_CAP*HDIM]        (18.9 MB)
#define OFF_BGU      42010752       // swizzled bf16 gate_up      (67 MB)
#define OFF_BDN      109119616      // swizzled bf16 down         (33.5 MB)
// total ~143 MB

// LDS A-stage: 64 rows x 64 k of bf16, row stride 144B (bank-conflict pad)
#define LDS_STRIDE 144
#define LDS_BYTES (64 * LDS_STRIDE)

// float -> bf16 pair, round-to-nearest via +0x8000, pack with one v_perm_b32
static __device__ __forceinline__ unsigned int pk_bf16(float lo, float hi) {
    unsigned int ul = __float_as_uint(lo) + 0x8000u;
    unsigned int uh = __float_as_uint(hi) + 0x8000u;
    return __builtin_amdgcn_perm(uh, ul, 0x07060302u);  // {uh[31:16], ul[31:16]}
}
static __device__ __forceinline__ unsigned short bf16_1(float f) {
    return (unsigned short)((__float_as_uint(f) + 0x8000u) >> 16);
}
static __device__ __forceinline__ v8f wmma_bf16(v16bf a, v16bf b, v8f c) {
    return __builtin_amdgcn_wmma_f32_16x16x32_bf16(false, a, false, b, (short)0, c,
                                                   false, false);
}
static __device__ __forceinline__ v16bf frag2(uint4 lo, uint4 hi) {
    v8u au;
    au[0] = lo.x; au[1] = lo.y; au[2] = lo.z; au[3] = lo.w;
    au[4] = hi.x; au[5] = hi.y; au[6] = hi.z; au[7] = hi.w;
    return __builtin_bit_cast(v16bf, au);
}
static __device__ __forceinline__ v16bf frag_from_lds(const char* p) {
    return frag2(*(const uint4*)p, *(const uint4*)(p + 32));
}

// ---------------- x -> bf16 (once per launch) ----------------
__global__ __launch_bounds__(256) void moe_cvt_x(const float* __restrict__ x,
                                                 unsigned short* __restrict__ xb) {
    int i = (blockIdx.x * 256 + threadIdx.x) * 8;
    float4 a = *(const float4*)(x + i);
    float4 b = *(const float4*)(x + i + 4);
    uint4 o;
    o.x = pk_bf16(a.x, a.y);
    o.y = pk_bf16(a.z, a.w);
    o.z = pk_bf16(b.x, b.y);
    o.w = pk_bf16(b.z, b.w);
    *(uint4*)(xb + i) = o;
}

// --------- weights -> WMMA-B-fragment-ready packed bf16 (once per launch) --------
// dst layout: [e][ntile(16 cols)][kblock(32 k)][lane(32)][2 x uint4]
// lane&15 = col-in-tile, lane>>4 = h; dword 4j+w holds (k=16j+8h+2w, +1)
__global__ __launch_bounds__(256) void moe_swz_b(const float* __restrict__ src,
                                                 uint4* __restrict__ dst, int N,
                                                 int kbBits, int ntBits) {
    unsigned int i = blockIdx.x * 256 + threadIdx.x;
    int vhalf = i & 1;
    int l = (i >> 1) & 31;
    int kb = (i >> 6) & ((1 << kbBits) - 1);
    int nt = (i >> (6 + kbBits)) & ((1 << ntBits) - 1);
    int e = i >> (6 + kbBits + ntBits);
    int col = nt * 16 + (l & 15);
    int kbase = kb * 32 + ((l >> 4) << 3) + (vhalf << 4);
    size_t K = (size_t)(1 << kbBits) * 32;
    const float* p = src + (size_t)e * K * N + (size_t)kbase * N + col;
    uint4 o;
    o.x = pk_bf16(p[0], p[(size_t)N]);
    o.y = pk_bf16(p[2 * (size_t)N], p[3 * (size_t)N]);
    o.z = pk_bf16(p[4 * (size_t)N], p[5 * (size_t)N]);
    o.w = pk_bf16(p[6 * (size_t)N], p[7 * (size_t)N]);
    dst[i] = o;
}

// ---------------- router ----------------
__global__ __launch_bounds__(256) void moe_router(
    const float* __restrict__ x, const float* __restrict__ rw,
    const float* __restrict__ rb, int* __restrict__ topk_idx,
    float* __restrict__ topk_w, int* __restrict__ counts) {
    const int t = blockIdx.x * 8 + (threadIdx.x >> 5);
    const int lane = threadIdx.x & 31;
    float acc[NEXP];
#pragma unroll
    for (int e = 0; e < NEXP; ++e) acc[e] = 0.f;
    const float* xr = x + (size_t)t * HDIM;
    for (int h = lane; h < HDIM; h += 32) {
        float xv = xr[h];
        const float* r = rw + h * NEXP;
#pragma unroll
        for (int e = 0; e < NEXP; ++e) acc[e] += xv * r[e];
    }
#pragma unroll
    for (int e = 0; e < NEXP; ++e) {
#pragma unroll
        for (int off = 16; off > 0; off >>= 1) acc[e] += __shfl_xor(acc[e], off, 32);
    }
    if (lane == 0) {
        float best0 = -1e30f, best1 = -1e30f;
        int i0 = 0, i1 = 0;
#pragma unroll
        for (int e = 0; e < NEXP; ++e) {
            float v = acc[e] + rb[e];
            if (v > best0) { best1 = best0; i1 = i0; best0 = v; i0 = e; }
            else if (v > best1) { best1 = v; i1 = e; }
        }
        float w0 = __builtin_amdgcn_rcpf(1.f + __expf(best1 - best0));
        topk_idx[t * 2 + 0] = i0;
        topk_idx[t * 2 + 1] = i1;
        topk_w[t * 2 + 0] = w0;
        topk_w[t * 2 + 1] = 1.f - w0;
        atomicAdd(&counts[i0], 1);
        atomicAdd(&counts[i1], 1);
    }
}

// ---------------- 64-aligned exclusive scan ----------------
__global__ void moe_scan(const int* __restrict__ counts, int* __restrict__ offs) {
    if (threadIdx.x == 0 && blockIdx.x == 0) {
        int off = 0;
#pragma unroll
        for (int e = 0; e < NEXP; ++e) {
            offs[e] = off;
            off += (counts[e] + 63) & ~63;
        }
        offs[NEXP] = off;
    }
}

// ---------------- scatter: (token,k) -> slot, remember the slot ----------------
__global__ __launch_bounds__(256) void moe_scatter(
    const int* __restrict__ topk_idx, const int* __restrict__ offs,
    int* __restrict__ counts2, int* __restrict__ etok, int* __restrict__ slot_of) {
    int i = blockIdx.x * blockDim.x + threadIdx.x;
    if (i >= S_TOK * 2) return;
    int e = topk_idx[i];
    int pos = atomicAdd(&counts2[e], 1);
    int slot = offs[e] + pos;
    etok[slot] = i >> 1;
    slot_of[i] = slot;
}

// ---------------- gate_up GEMM (64 rows x 128 act cols / block) ----------------
__global__ __launch_bounds__(256) void moe_gate_up(
    const unsigned short* __restrict__ xb, const uint4* __restrict__ bswz,
    const float* __restrict__ gub, const int* __restrict__ offs,
    const int* __restrict__ etok, unsigned short* __restrict__ act) {
    __shared__ __align__(16) unsigned char smem[LDS_BYTES];
    const int tid = threadIdx.x;
    const int wave = tid >> 5, lane = tid & 31;
    const int slot_base = blockIdx.x * 64;
    if (slot_base >= offs[NEXP]) return;
    int e = 0;
#pragma unroll
    for (int i = 1; i < NEXP; ++i)
        if (slot_base >= offs[i]) e = i;

    // staging role: thread -> (row, quad); each thread stores 2 x 16B per chunk
    const int srow = tid >> 2, sq = tid & 3;
    int stok = etok[slot_base + srow];
    const unsigned short* srcrow = xb + (size_t)(stok < 0 ? 0 : stok) * HDIM + sq * 8;
    char* sdst = (char*)smem + srow * LDS_STRIDE + (sq << 4);

    const int nt_g = blockIdx.y * 8 + wave;                       // gate tile id
    const int colg = nt_g * 16 + (lane & 15);                     // act/gate col
    const uint4* bg4 = bswz + ((size_t)e * 256 + nt_g) * 2048 + lane * 2;
    const uint4* bu4 = bg4 + (size_t)128 * 2048;                  // up tiles
    const char* lbase = (const char*)smem + (lane & 15) * LDS_STRIDE + ((lane >> 4) << 4);

    v8f accg[4] = {}, accu[4] = {};
    for (int kb2 = 0; kb2 < HDIM / 64; ++kb2) {
        __syncthreads();
        *(uint4*)(sdst) = *(const uint4*)(srcrow + kb2 * 64);
        *(uint4*)(sdst + 64) = *(const uint4*)(srcrow + kb2 * 64 + 32);
        __syncthreads();
#pragma unroll
        for (int half = 0; half < 2; ++half) {
            const int kb = kb2 * 2 + half;
            uint4 glo = bg4[kb * 64], ghi = bg4[kb * 64 + 1];
            uint4 ulo = bu4[kb * 64], uhi = bu4[kb * 64 + 1];
            v16bf g = frag2(glo, ghi);
            v16bf u = frag2(ulo, uhi);
#pragma unroll
            for (int s = 0; s < 4; ++s) {
                v16bf a = frag_from_lds(lbase + s * (16 * LDS_STRIDE) + half * 64);
                accg[s] = wmma_bf16(a, g, accg[s]);
                accu[s] = wmma_bf16(a, u, accu[s]);
            }
        }
    }

    const float biasg = gub[(size_t)e * TWOI + colg];
    const float biasu = gub[(size_t)e * TWOI + INTER + colg];
#pragma unroll
    for (int s = 0; s < 4; ++s) {
#pragma unroll
        for (int r = 0; r < 8; ++r) {
            int m = s * 16 + ((lane >> 4) << 3) + r;
            float g = fminf(accg[s][r] + biasg, LIMIT);
            float u = fminf(fmaxf(accu[s][r] + biasu, -LIMIT), LIMIT);
            float glu = g * __builtin_amdgcn_rcpf(1.f + __expf(-ALPHA * g));
            act[(size_t)(slot_base + m) * INTER + colg] = bf16_1((u + 1.f) * glu);
        }
    }
}

// ---------------- down GEMM -> per-slot rows (no atomics) ----------------
__global__ __launch_bounds__(256) void moe_down(
    const unsigned short* __restrict__ act, const uint4* __restrict__ bswz,
    const float* __restrict__ db, const int* __restrict__ offs,
    float* __restrict__ dout) {
    __shared__ __align__(16) unsigned char smem[LDS_BYTES];
    const int tid = threadIdx.x;
    const int wave = tid >> 5, lane = tid & 31;
    const int slot_base = blockIdx.x * 64;
    if (slot_base >= offs[NEXP]) return;
    int e = 0;
#pragma unroll
    for (int i = 1; i < NEXP; ++i)
        if (slot_base >= offs[i]) e = i;

    const int srow = tid >> 2, sq = tid & 3;
    const unsigned short* srcrow = act + (size_t)(slot_base + srow) * INTER + sq * 8;
    char* sdst = (char*)smem + srow * LDS_STRIDE + (sq << 4);

    const int nt = blockIdx.y * 8 + wave;
    const int hcol = nt * 16 + (lane & 15);
    const uint4* bd4 = bswz + ((size_t)e * 64 + nt) * 4096 + lane * 2;
    const char* lbase = (const char*)smem + (lane & 15) * LDS_STRIDE + ((lane >> 4) << 4);

    v8f acc[4] = {};
    for (int kb2 = 0; kb2 < INTER / 64; ++kb2) {
        __syncthreads();
        *(uint4*)(sdst) = *(const uint4*)(srcrow + kb2 * 64);
        *(uint4*)(sdst + 64) = *(const uint4*)(srcrow + kb2 * 64 + 32);
        __syncthreads();
#pragma unroll
        for (int half = 0; half < 2; ++half) {
            const int kb = kb2 * 2 + half;
            v16bf b = frag2(bd4[kb * 64], bd4[kb * 64 + 1]);
#pragma unroll
            for (int s = 0; s < 4; ++s) {
                v16bf a = frag_from_lds(lbase + s * (16 * LDS_STRIDE) + half * 64);
                acc[s] = wmma_bf16(a, b, acc[s]);
            }
        }
    }

    const float bias = db[(size_t)e * HDIM + hcol];
#pragma unroll
    for (int s = 0; s < 4; ++s) {
#pragma unroll
        for (int r = 0; r < 8; ++r) {
            int m = s * 16 + ((lane >> 4) << 3) + r;
            dout[(size_t)(slot_base + m) * HDIM + hcol] = acc[s][r] + bias;
        }
    }
}

// ---------------- deterministic combine: out = w0*row0 + w1*row1 ----------------
__global__ __launch_bounds__(256) void moe_combine(
    const float* __restrict__ dout, const float* __restrict__ topk_w,
    const int* __restrict__ slot_of, float* __restrict__ out) {
    const int t = blockIdx.x;
    const int h = threadIdx.x * 4;
    const int s0 = slot_of[t * 2 + 0];
    const int s1 = slot_of[t * 2 + 1];
    const float w0 = topk_w[t * 2 + 0];
    const float w1 = topk_w[t * 2 + 1];
    float4 a = *(const float4*)(dout + (size_t)s0 * HDIM + h);
    float4 b = *(const float4*)(dout + (size_t)s1 * HDIM + h);
    float4 o;
    o.x = w0 * a.x + w1 * b.x;
    o.y = w0 * a.y + w1 * b.y;
    o.z = w0 * a.z + w1 * b.z;
    o.w = w0 * a.w + w1 * b.w;
    *(float4*)(out + (size_t)t * HDIM + h) = o;
}

extern "C" void kernel_launch(void* const* d_in, const int* in_sizes, int n_in,
                              void* d_out, int out_size, void* d_ws, size_t ws_size,
                              hipStream_t stream) {
    const float* x   = (const float*)d_in[0];
    const float* rw  = (const float*)d_in[1];
    const float* rb  = (const float*)d_in[2];
    const float* gup = (const float*)d_in[3];
    const float* gub = (const float*)d_in[4];
    const float* dp  = (const float*)d_in[5];
    const float* db  = (const float*)d_in[6];
    float* out = (float*)d_out;
    char* ws = (char*)d_ws;

    int*   topk_idx = (int*)(ws + OFF_TOPK_IDX);
    float* topk_w   = (float*)(ws + OFF_TOPK_W);
    int*   slot_of  = (int*)(ws + OFF_SLOTOF);
    int*   counts   = (int*)(ws + OFF_COUNTS);
    int*   counts2  = (int*)(ws + OFF_COUNTS2);
    int*   offs     = (int*)(ws + OFF_OFFS);
    int*   etok     = (int*)(ws + OFF_ETOK);
    unsigned short* xb   = (unsigned short*)(ws + OFF_XB);
    unsigned short* actb = (unsigned short*)(ws + OFF_ACT);
    float* doutb    = (float*)(ws + OFF_DOUT);
    uint4* bgu      = (uint4*)(ws + OFF_BGU);
    uint4* bdn      = (uint4*)(ws + OFF_BDN);

    hipMemsetAsync(ws + OFF_COUNTS, 0, 128, stream);            // counts, counts2, offs
    hipMemsetAsync(ws + OFF_ETOK, 0xFF, SLOT_CAP * 4, stream);  // slot tokens = -1

    // one-pass conversions/swizzles (HBM-bound, ~13us total at 23.3 TB/s)
    moe_cvt_x<<<(S_TOK * HDIM) / (256 * 8), 256, 0, stream>>>(x, xb);
    moe_swz_b<<<(NEXP * 256 * 32 * 32 * 2) / 256, 256, 0, stream>>>(gup, bgu, TWOI, 5, 8);
    moe_swz_b<<<(NEXP * 64 * 64 * 32 * 2) / 256, 256, 0, stream>>>(dp, bdn, HDIM, 6, 6);

    moe_router<<<S_TOK / 8, 256, 0, stream>>>(x, rw, rb, topk_idx, topk_w, counts);
    moe_scan<<<1, 32, 0, stream>>>(counts, offs);
    moe_scatter<<<(S_TOK * 2 + 255) / 256, 256, 0, stream>>>(topk_idx, offs, counts2,
                                                             etok, slot_of);
    moe_gate_up<<<dim3(TILES_M, TWOI / 256), 256, 0, stream>>>(xb, bgu, gub, offs,
                                                               etok, actb);
    moe_down<<<dim3(TILES_M, HDIM / 128), 256, 0, stream>>>(actb, bdn, db, offs, doutb);
    moe_combine<<<S_TOK, 256, 0, stream>>>(doutb, topk_w, slot_of, out);
}
